// GNNLinkPredictor_19078244729525
// MI455X (gfx1250) — compile-verified, matrix operand order checked by reading, not run on hardware
//
#include <hip/hip_runtime.h>
#include <hip/hip_bf16.h>

typedef __attribute__((ext_vector_type(16))) _Float16 v16h;
typedef __attribute__((ext_vector_type(8)))  float    v8f;

#define NNODES 100000
#define NEDGES 1600000
#define NLABEL 200000

// ---------------------------------------------------------------- zero
__global__ void gnn_zero_kernel(float4* __restrict__ p, long n4) {
    long i = (long)blockIdx.x * blockDim.x + threadIdx.x;
    if (i < n4) p[i] = make_float4(0.f, 0.f, 0.f, 0.f);
}

// ---------------------------------------------------------------- degree
__global__ void gnn_degree_kernel(const int* __restrict__ dst, float* __restrict__ deg, int nedges) {
    int e = blockIdx.x * blockDim.x + threadIdx.x;
    if (e < nedges) atomicAdd(&deg[dst[e]], 1.0f);
}

__global__ void gnn_invdeg_kernel(float* __restrict__ deg, int n) {
    int i = blockIdx.x * blockDim.x + threadIdx.x;
    if (i < n) deg[i] = 1.0f / fmaxf(deg[i], 1.0f);
}

// ---------------------------------------------------------------- scatter-add (mean aggregation numerator)
template <int FEAT>
__global__ void gnn_scatter_kernel(const float* __restrict__ x,
                                   const int* __restrict__ src,
                                   const int* __restrict__ dst,
                                   float* __restrict__ agg, int nedges) {
    constexpr int TPE = FEAT / 4;               // threads per edge (float4 chunks)
    long idx = (long)blockIdx.x * blockDim.x + threadIdx.x;
    long total = (long)nedges * TPE;
    if (idx >= total) return;
    int e = (int)(idx / TPE);
    int c = (int)(idx % TPE) * 4;
    int s = src[e], d = dst[e];
    float4 v = *(const float4*)(x + (long)s * FEAT + c);
    float* p = agg + (long)d * FEAT + c;
    atomicAdd(p + 0, v.x);
    atomicAdd(p + 1, v.y);
    atomicAdd(p + 2, v.z);
    atomicAdd(p + 3, v.w);
}

// ---------------------------------------------------------------- WMMA SAGE layer
// out[row][col] = act( sum_k (xa[row][k]*invd[row]) * wl[col][k]
//                    + sum_k  xb[row][k]            * wr[col][k]  + bias[col] )
// Concatenated K = 2*KSEG, tiled 16x16 per wave, K step = 32 (f16 WMMA).
template <int KSEG, int NCOLS, bool RELU>
__global__ void gnn_sage_wmma_kernel(const float* __restrict__ xa,
                                     const float* __restrict__ xb,
                                     const float* __restrict__ invd,
                                     const float* __restrict__ wl,
                                     const float* __restrict__ wr,
                                     const float* __restrict__ bias,
                                     float* __restrict__ out) {
    const int wave = threadIdx.x >> 5;
    const int lane = threadIdx.x & 31;
    const int half = lane >> 4;          // lanes 16-31 hold the other K sub-block
    const int r    = lane & 15;
    const int row  = blockIdx.x * 16 + r;  // A: lane -> matrix row (M = r, both halves)
    const int col  = wave * 16 + r;        // B: lane -> matrix column (N = r)
    const float sc = invd[row];

    union { v8f v; float f[8]; } acc;
#pragma unroll
    for (int i = 0; i < 8; ++i) acc.f[i] = 0.0f;

#pragma unroll
    for (int ks = 0; ks < 2 * KSEG; ks += 32) {
        union { v16h v; _Float16 e[16]; } A, B;
#pragma unroll
        for (int g = 0; g < 2; ++g) {
            // ISA 16-bit A layout: vgpr pair group g covers K = ks + g*16 + half*8 .. +7
            const int kk = ks + g * 16 + half * 8;
            // ---- A operand (activations, f32 -> f16 with inv_deg scaling on first half of K)
            {
                const float* s;
                float m;
                if (kk < KSEG) { s = xa + (long)row * KSEG + kk;         m = sc;   }
                else           { s = xb + (long)row * KSEG + (kk - KSEG); m = 1.0f; }
                float4 p0 = *(const float4*)(s);
                float4 p1 = *(const float4*)(s + 4);
                A.e[g * 8 + 0] = (_Float16)(p0.x * m);
                A.e[g * 8 + 1] = (_Float16)(p0.y * m);
                A.e[g * 8 + 2] = (_Float16)(p0.z * m);
                A.e[g * 8 + 3] = (_Float16)(p0.w * m);
                A.e[g * 8 + 4] = (_Float16)(p1.x * m);
                A.e[g * 8 + 5] = (_Float16)(p1.y * m);
                A.e[g * 8 + 6] = (_Float16)(p1.z * m);
                A.e[g * 8 + 7] = (_Float16)(p1.w * m);
            }
            // ---- B operand (weights, [NCOLS][KSEG] row-major; column n = r)
            {
                const float* s;
                if (kk < KSEG) s = wl + (long)col * KSEG + kk;
                else           s = wr + (long)col * KSEG + (kk - KSEG);
                float4 p0 = *(const float4*)(s);
                float4 p1 = *(const float4*)(s + 4);
                B.e[g * 8 + 0] = (_Float16)p0.x;
                B.e[g * 8 + 1] = (_Float16)p0.y;
                B.e[g * 8 + 2] = (_Float16)p0.z;
                B.e[g * 8 + 3] = (_Float16)p0.w;
                B.e[g * 8 + 4] = (_Float16)p1.x;
                B.e[g * 8 + 5] = (_Float16)p1.y;
                B.e[g * 8 + 6] = (_Float16)p1.z;
                B.e[g * 8 + 7] = (_Float16)p1.w;
            }
        }
        acc.v = __builtin_amdgcn_wmma_f32_16x16x32_f16(
            /*neg_a=*/false, A.v, /*neg_b=*/false, B.v,
            /*c_mod=*/(short)0, acc.v, /*reuse_a=*/false, /*reuse_b=*/false);
    }

    const float bv = bias[col];
#pragma unroll
    for (int i = 0; i < 8; ++i) {
        int m = i + half * 8;              // C/D layout: VGPR i -> row i (+8 for upper half)
        float v = acc.f[i] + bv;
        if (RELU) v = fmaxf(v, 0.0f);
        out[((long)blockIdx.x * 16 + m) * NCOLS + col] = v;
    }
}

// ---------------------------------------------------------------- decode: dot(z[i0], z[i1])
__global__ void gnn_decode_kernel(const float* __restrict__ z,
                                  const int* __restrict__ i0,
                                  const int* __restrict__ i1,
                                  float* __restrict__ out, int n) {
    int t = blockIdx.x * blockDim.x + threadIdx.x;
    int pair = t >> 4;
    int l = t & 15;                        // 16 lanes x float4 = 64 features
    if (pair >= n) return;
    long a = i0[pair], b = i1[pair];
    float4 va = *(const float4*)(z + a * 64 + l * 4);
    float4 vb = *(const float4*)(z + b * 64 + l * 4);
    float s = va.x * vb.x + va.y * vb.y + va.z * vb.z + va.w * vb.w;
#pragma unroll
    for (int off = 8; off > 0; off >>= 1) s += __shfl_down(s, off, 16);
    if (l == 0) out[pair] = s;
}

// ---------------------------------------------------------------- launcher
extern "C" void kernel_launch(void* const* d_in, const int* in_sizes, int n_in,
                              void* d_out, int out_size, void* d_ws, size_t ws_size,
                              hipStream_t stream) {
    const float* emb = (const float*)d_in[0];
    const float* w1l = (const float*)d_in[1];
    const float* w1r = (const float*)d_in[2];
    const float* b1  = (const float*)d_in[3];
    const float* w2l = (const float*)d_in[4];
    const float* w2r = (const float*)d_in[5];
    const float* b2  = (const float*)d_in[6];
    const int*   ei  = (const int*)d_in[7];     // [2, NEDGES] row-major
    const int*   eli = (const int*)d_in[8];     // [2, NLABEL] row-major
    float* outp = (float*)d_out;

    const int* esrc = ei;
    const int* edst = ei + NEDGES;
    const int* li0  = eli;
    const int* li1  = eli + NLABEL;

    // workspace layout (floats)
    float* ws   = (float*)d_ws;
    float* deg  = ws;                              // NNODES        -> becomes inv_deg
    float* agg1 = deg  + NNODES;                   // NNODES*64
    float* h    = agg1 + (long)NNODES * 64;        // NNODES*128
    float* agg2 = h    + (long)NNODES * 128;       // NNODES*128
    float* z    = agg2 + (long)NNODES * 128;       // NNODES*64

    // 1) zero deg+agg1 (contiguous) and agg2 — required every call (atomic accumulators)
    {
        long n4 = ((long)NNODES + (long)NNODES * 64) / 4;
        gnn_zero_kernel<<<(int)((n4 + 255) / 256), 256, 0, stream>>>((float4*)deg, n4);
        long m4 = ((long)NNODES * 128) / 4;
        gnn_zero_kernel<<<(int)((m4 + 255) / 256), 256, 0, stream>>>((float4*)agg2, m4);
    }

    // 2) in-degree, then inv_deg = 1/max(deg,1)
    gnn_degree_kernel<<<(NEDGES + 255) / 256, 256, 0, stream>>>(edst, deg, NEDGES);
    gnn_invdeg_kernel<<<(NNODES + 255) / 256, 256, 0, stream>>>(deg, NNODES);

    // 3) layer-1 aggregation: agg1 += emb[src] scattered to dst
    {
        long total = (long)NEDGES * 16;            // 64/4 chunks per edge
        gnn_scatter_kernel<64><<<(int)((total + 255) / 256), 256, 0, stream>>>(
            emb, esrc, edst, agg1, NEDGES);
    }

    // 4) h = relu([agg1*invd | emb] @ [w1l|w1r]^T + b1)   (K=128, 128 cols, 8 waves/block)
    gnn_sage_wmma_kernel<64, 128, true><<<NNODES / 16, 256, 0, stream>>>(
        agg1, emb, deg, w1l, w1r, b1, h);

    // 5) layer-2 aggregation: agg2 += h[src]
    {
        long total = (long)NEDGES * 32;            // 128/4 chunks per edge
        gnn_scatter_kernel<128><<<(int)((total + 255) / 256), 256, 0, stream>>>(
            h, esrc, edst, agg2, NEDGES);
    }

    // 6) z = [agg2*invd | h] @ [w2l|w2r]^T + b2           (K=256, 64 cols, 4 waves/block)
    gnn_sage_wmma_kernel<128, 64, false><<<NNODES / 16, 128, 0, stream>>>(
        agg2, h, deg, w2l, w2r, b2, z);

    // 7) decode: out[k] = dot(z[li0[k]], z[li1[k]])
    {
        long total = (long)NLABEL * 16;
        gnn_decode_kernel<<<(int)((total + 255) / 256), 256, 0, stream>>>(
            z, li0, li1, outp, NLABEL);
    }
}
